// BertIntermediate_54726473285716
// MI455X (gfx1250) — 8-bit matrix operand order repaired and checked by reading, compiled, not run on hardware
//
#include <hip/hip_runtime.h>

typedef __attribute__((ext_vector_type(8))) int v8i;

__device__ __forceinline__ signed char fq1(float v, float inv) {
    float r = rintf(v * inv);
    r = fminf(fmaxf(r, -128.0f), 127.0f);
    return (signed char)(int)r;
}

__device__ __forceinline__ v8i loadA(const signed char* p) {
    int2 a0 = *(const int2*)(p);
    int2 a1 = *(const int2*)(p + 8);
    int2 a2 = *(const int2*)(p + 32);
    int2 a3 = *(const int2*)(p + 40);
    v8i r = {a0.x, a0.y, a1.x, a1.y, a2.x, a2.y, a3.x, a3.y};
    return r;
}

__device__ __forceinline__ v8i loadB(const signed char* p) {
    int4 lo = *(const int4*)(p);
    int4 hi = *(const int4*)(p + 32);
    v8i r = {lo.x, lo.y, lo.z, lo.w, hi.x, hi.y, hi.z, hi.w};
    return r;
}

__global__ void init_amax(int* amax) {
    if (threadIdx.x < 2) amax[threadIdx.x] = 0;
}

__global__ void amax_abs4(const float4* __restrict__ v, long n4,
                          int* __restrict__ out) {
    const long stride = (long)gridDim.x * blockDim.x;
    float m = 0.0f;
    for (long i = (long)blockIdx.x * blockDim.x + threadIdx.x; i < n4; i += stride) {
        float4 f = v[i];
        m = fmaxf(m, fmaxf(fmaxf(fabsf(f.x), fabsf(f.y)),
                           fmaxf(fabsf(f.z), fabsf(f.w))));
    }
#pragma unroll
    for (int off = 16; off > 0; off >>= 1)
        m = fmaxf(m, __shfl_xor(m, off, 32));
    if ((threadIdx.x & 31) == 0)
        atomicMax(out, __float_as_int(m));
}

__global__ void quantize4(const float* __restrict__ v, signed char* __restrict__ q,
                          long n4, const int* __restrict__ amaxBits) {
    const float inv = 127.0f / __int_as_float(*amaxBits);
    long i = (long)blockIdx.x * blockDim.x + threadIdx.x;
    if (i >= n4) return;
    float4 f = ((const float4*)v)[i];
    char4 c;
    c.x = fq1(f.x, inv);
    c.y = fq1(f.y, inv);
    c.z = fq1(f.z, inv);
    c.w = fq1(f.w, inv);
    ((char4*)q)[i] = c;
}

__global__ __launch_bounds__(256)
void i8gemm_bias_gelu(const signed char* __restrict__ xq,
                      const signed char* __restrict__ wq,
                      const float* __restrict__ bias,
                      const int* __restrict__ amaxBits,
                      float* __restrict__ out,
                      int Kdim, int Ndim) {
    const int wave  = threadIdx.x >> 5;
    const int lane  = threadIdx.x & 31;
    const int half  = lane >> 4;
    const int l16   = lane & 15;
    const int waveM = wave & 3;
    const int waveN = wave >> 2;

    const long row0 = (long)blockIdx.y * 128 + waveM * 32;
    const int  col0 = blockIdx.x * 128 + waveN * 64;

    const signed char* aPtr0 = xq + (row0 + l16) * (long)Kdim + half * 16;
    const signed char* aPtr1 = aPtr0 + 16 * (long)Kdim;

    const signed char* bPtr[4];
    bPtr[0] = wq + (long)(col0 + l16) * Kdim + half * 16;
    bPtr[1] = bPtr[0] + 16 * (long)Kdim;
    bPtr[2] = bPtr[0] + 32 * (long)Kdim;
    bPtr[3] = bPtr[0] + 48 * (long)Kdim;

    v8i acc[2][4];
#pragma unroll
    for (int i = 0; i < 2; ++i)
#pragma unroll
        for (int t = 0; t < 4; ++t) {
            v8i z = {};
            acc[i][t] = z;
        }

    for (int k0 = 0; k0 < Kdim; k0 += 64) {
        v8i A0 = loadA(aPtr0 + k0);
        v8i A1 = loadA(aPtr1 + k0);
#pragma unroll
        for (int t = 0; t < 4; ++t) {
            v8i B = loadB(bPtr[t] + k0);
            acc[0][t] = __builtin_amdgcn_wmma_i32_16x16x64_iu8(
                true, A0, true, B, acc[0][t], false, false);
            acc[1][t] = __builtin_amdgcn_wmma_i32_16x16x64_iu8(
                true, A1, true, B, acc[1][t], false, false);
        }
    }

    const float s = (__int_as_float(amaxBits[0]) * (1.0f / 127.0f)) *
                    (__int_as_float(amaxBits[1]) * (1.0f / 127.0f));

#pragma unroll
    for (int i = 0; i < 2; ++i) {
#pragma unroll
        for (int t = 0; t < 4; ++t) {
            const int   n  = col0 + t * 16 + l16;
            const float bv = bias[n];
            const long  rb = row0 + i * 16;
#pragma unroll
            for (int r = 0; r < 8; ++r) {
                const int m = half * 8 + r;
                float h = (float)acc[i][t][r] * s + bv;
                float g = 0.5f * h * (1.0f + erff(h * 0.70710678118654752f));
                out[(rb + m) * (long)Ndim + (long)n] = g;
            }
        }
    }
}

extern "C" void kernel_launch(void* const* d_in, const int* in_sizes, int n_in,
                              void* d_out, int out_size, void* d_ws, size_t ws_size,
                              hipStream_t stream) {
    const float* x = (const float*)d_in[0];
    const float* W = (const float*)d_in[1];
    const float* b = (const float*)d_in[2];
    float* out = (float*)d_out;

    const int  N = in_sizes[2];
    const int  K = in_sizes[1] / N;
    const long M = (long)in_sizes[0] / K;

    int*         amax = (int*)d_ws;
    signed char* xq   = (signed char*)d_ws + 256;
    signed char* wq   = xq + (size_t)M * K;

    hipLaunchKernelGGL(init_amax, dim3(1), dim3(32), 0, stream, amax);

    const long nx4 = (long)in_sizes[0] / 4;
    const long nw4 = (long)in_sizes[1] / 4;
    hipLaunchKernelGGL(amax_abs4, dim3(1024), dim3(256), 0, stream,
                       (const float4*)x, nx4, amax + 0);
    hipLaunchKernelGGL(amax_abs4, dim3(1024), dim3(256), 0, stream,
                       (const float4*)W, nw4, amax + 1);

    hipLaunchKernelGGL(quantize4, dim3((unsigned)((nx4 + 255) / 256)), dim3(256), 0,
                       stream, x, xq, nx4, amax + 0);
    hipLaunchKernelGGL(quantize4, dim3((unsigned)((nw4 + 255) / 256)), dim3(256), 0,
                       stream, W, wq, nw4, amax + 1);

    dim3 grid((unsigned)(N / 128), (unsigned)(M / 128));
    hipLaunchKernelGGL(i8gemm_bias_gelu, grid, dim3(256), 0, stream,
                       xq, wq, b, amax, out, K, N);
}
